// Waveletnet_45277545234505
// MI455X (gfx1250) — matrix operand order repaired and checked by reading, not compiled, not run on hardware
//
#include <hip/hip_runtime.h>
#include <hip/hip_bf16.h>

// ---------------------------------------------------------------------------
// Wavelet U-Net on MI455X (gfx1250, wave32).
// conv3x3 = implicit GEMM on V_WMMA_F32_16X16X32_F16 (f32 accum).
// Block tile: M=64 spatial x N=128 out-channels, K chunk = 32 over Cin*9,
// K ordered as f*Cin+ci (f = 3x3 tap). 8 waves (2x4); each wave owns a
// 32x32 output quadrant = 4 WMMA subtiles fed by 2 A + 2 B fragments
// (8 x ds_load_b128). LDS is double-buffered: next chunk staged while the
// current chunk's WMMAs run -> one barrier per chunk.
// FASTK (Cin >= 32): a 32-K chunk never crosses a tap boundary, so all tap /
// border math is chunk-uniform and hoisted; A addresses become base + j*step.
// ---------------------------------------------------------------------------

typedef __attribute__((ext_vector_type(16))) _Float16 v16h;
typedef __attribute__((ext_vector_type(8)))  float    v8f;

// ------------- fp32 [Cout][Cin][3][3] -> f16 [Cout][f][Cin] ----------------
__global__ __launch_bounds__(256) void cvt_w_kernel(const float* __restrict__ in,
                                                    _Float16* __restrict__ out,
                                                    int Cin, int n) {
    int i = blockIdx.x * 256 + threadIdx.x;
    if (i >= n) return;
    int cin9 = Cin * 9;
    int co = i / cin9;
    int r  = i - co * cin9;
    int ci = r / 9;
    int f  = r - ci * 9;
    out[(long long)co * cin9 + f * Cin + ci] = (_Float16)in[i];
}

// ---------------- Haar forward wavelet (wt) --------------------------------
__global__ __launch_bounds__(256) void wt_kernel(const float* __restrict__ in,
                                                 long long in_bstride, int in_coff,
                                                 float* __restrict__ out,
                                                 int lC, int lw, int n) {
    int idx = blockIdx.x * 256 + threadIdx.x;
    if (idx >= n) return;
    const int Wo = 1 << lw;
    int w = idx & (Wo - 1);
    int h = (idx >> lw) & (Wo - 1);
    int c = (idx >> (2 * lw)) & ((1 << lC) - 1);
    int b = idx >> (2 * lw + lC);

    long long plane_in = (long long)4 * Wo * Wo;
    const float* src = in + (long long)b * in_bstride + (long long)(in_coff + c) * plane_in
                          + (long long)(2 * h) * (2 * Wo) + 2 * w;
    float s00 = src[0], s01 = src[1];
    float s10 = src[2 * Wo], s11 = src[2 * Wo + 1];

    float y0 = 0.25f * (s00 + s01 + s10 + s11);
    float y1 = 0.25f * (s00 + s01 - s10 - s11) + 0.5f;
    float y2 = 0.25f * (s00 - s01 + s10 - s11) + 0.5f;
    float y3 = 0.25f * (s00 - s01 - s10 + s11) + 0.5f;

    long long po = (long long)Wo * Wo;
    float* dst = out + ((((long long)b << lC) + c) * 4) * po + (long long)h * Wo + w;
    dst[0]      = y0;
    dst[po]     = y1;
    dst[2 * po] = y2;
    dst[3 * po] = y3;
}

// ---------------- Haar inverse wavelet (iwt), optional sigmoid -------------
template<bool SIG>
__global__ __launch_bounds__(256) void iwt_kernel(const float* __restrict__ in,
                                                  float* __restrict__ out,
                                                  long long out_bstride, int out_coff,
                                                  int lC, int lw, int n) {
    int idx = blockIdx.x * 256 + threadIdx.x;
    if (idx >= n) return;
    const int Wi = 1 << lw;
    int w = idx & (Wi - 1);
    int h = (idx >> lw) & (Wi - 1);
    int c = (idx >> (2 * lw)) & ((1 << lC) - 1);
    int b = idx >> (2 * lw + lC);

    long long plane = (long long)Wi * Wi;
    const float* src = in + ((((long long)b << lC) + c) * 4) * plane + (long long)h * Wi + w;
    float z0 = src[0];
    float z1 = 2.f * src[plane]     - 1.f;
    float z2 = 2.f * src[2 * plane] - 1.f;
    float z3 = 2.f * src[3 * plane] - 1.f;

    float x00 = z0 + 0.5f * ( z1 + z2 + z3);
    float x01 = z0 + 0.5f * ( z1 - z2 - z3);
    float x10 = z0 + 0.5f * (-z1 + z2 - z3);
    float x11 = z0 + 0.5f * (-z1 - z2 + z3);
    if (SIG) {
        x00 = 1.f / (1.f + __expf(-x00));
        x01 = 1.f / (1.f + __expf(-x01));
        x10 = 1.f / (1.f + __expf(-x10));
        x11 = 1.f / (1.f + __expf(-x11));
    }
    float* dst = out + (long long)b * out_bstride + (long long)(out_coff + c) * (4 * plane)
                     + (long long)(2 * h) * (2 * Wi) + 2 * w;
    dst[0]          = x00;
    dst[1]          = x01;
    dst[2 * Wi]     = x10;
    dst[2 * Wi + 1] = x11;
}

// ---------------- conv3x3 (pad 1) implicit GEMM via WMMA -------------------
// in : [B][Cin][W][W] at b*in_bstride (W = 1<<lw, Cin = 1<<lcin)
// wgt: [Cout][9<<lcin] f16, K = f*Cin + ci; bias [Cout]
// out: [B][..][W][W] at channel out_coff+n
template<bool HAS_RES, bool FASTK>
__global__ __launch_bounds__(256)
void conv3x3_wmma_kernel(const float* __restrict__ in, long long in_bstride,
                         const _Float16* __restrict__ wgt, const float* __restrict__ bias,
                         const float* __restrict__ res,
                         float* __restrict__ out, long long out_bstride, int out_coff,
                         int lcin, int Cout, int lw) {
    __shared__ _Float16 Als[2][64][40];     // [buf][m][k]
    __shared__ _Float16 Bls[2][128][40];    // [buf][n][k] (K-contiguous -> b128)

    const int tid  = threadIdx.x;
    const int b    = blockIdx.z;
    const int m0   = blockIdx.x << 6;       // 64 spatial positions
    const int n0   = blockIdx.y << 7;       // 128 output channels
    const int Cin  = 1 << lcin;
    const int Ktot = 9 << lcin;
    const int W    = 1 << lw;
    const int HW   = 1 << (2 * lw);

    const int wv   = tid >> 5, lane = tid & 31;
    const int half = lane >> 4, l16 = lane & 15;
    const int mw   = (wv >> 2) << 5;        // 0 / 32
    const int nw   = (wv & 3) << 5;         // 0..96

    const float* inb = in + (long long)b * in_bstride;

    // A staging: thread owns spatial row m = tid&63, stages k = (tid>>6) + 4j
    const int am   = tid & 63;
    const int ak   = tid >> 6;              // 0..3
    const int apos = m0 + am;
    const int aoh  = apos >> lw, aow = apos & (W - 1);
    // B staging: thread owns k = tid&31, stages n = (tid>>5) + 8j
    const int bk   = tid & 31;
    const int bnb  = tid >> 5;              // 0..7

    auto stageA = [&](int buf, int k0) {
        if (FASTK) {
            // chunk-uniform tap: f scalar, border math hoisted out of j-loop
            int f  = k0 >> lcin;
            int fy = (f * 11) >> 5, fx = f - 3 * fy;
            int ih = aoh + fy - 1, iw = aow + fx - 1;
            bool ok = ((unsigned)ih < (unsigned)W) & ((unsigned)iw < (unsigned)W);
            int ihc = ih < 0 ? 0 : (ih >= W ? W - 1 : ih);
            int iwc = iw < 0 ? 0 : (iw >= W ? W - 1 : iw);
            int base = ((((k0 & (Cin - 1)) + ak) << (2 * lw))) + (ihc << lw) + iwc;
            #pragma unroll
            for (int j = 0; j < 8; ++j) {
                float v = inb[base + ((4 * j) << (2 * lw))];
                Als[buf][am][ak + 4 * j] = (_Float16)(ok ? v : 0.f);
            }
        } else {
            #pragma unroll
            for (int j = 0; j < 8; ++j) {
                int k  = ak + 4 * j;
                int kk = k0 + k;
                int f9 = kk >> lcin;
                int f  = f9 > 8 ? 8 : f9;
                int fy = (f * 11) >> 5;
                int fx = f - 3 * fy;
                int ci = kk & (Cin - 1);
                int ih = aoh + fy - 1, iw = aow + fx - 1;
                bool ok = (f9 < 9) & ((unsigned)ih < (unsigned)W) & ((unsigned)iw < (unsigned)W);
                int ihc = ih < 0 ? 0 : (ih >= W ? W - 1 : ih);
                int iwc = iw < 0 ? 0 : (iw >= W ? W - 1 : iw);
                float v = inb[(ci << (2 * lw)) + (ihc << lw) + iwc];
                Als[buf][am][k] = (_Float16)(ok ? v : 0.f);
            }
        }
    };
    auto stageB = [&](int buf, int k0) {
        int kk = k0 + bk;
        bool okk = FASTK ? true : (kk < Ktot);     // Ktot % 32 == 0 when Cin>=32
        int kkc = okk ? kk : Ktot - 1;
        #pragma unroll
        for (int j = 0; j < 16; ++j) {
            int n   = n0 + bnb + 8 * j;
            int ncl = n < Cout ? n : Cout - 1;
            _Float16 w_ = wgt[(long long)ncl * Ktot + kkc];
            Bls[buf][bnb + 8 * j][bk] = (okk && n < Cout) ? w_ : (_Float16)0.f;
        }
    };

    v8f acc00 = {}, acc01 = {}, acc10 = {}, acc11 = {};

    stageA(0, 0);
    stageB(0, 0);
    __syncthreads();

    for (int k0 = 0; k0 < Ktot; k0 += 32) {
        const int cur = (k0 >> 5) & 1;
        const int k1  = k0 + 32;
        if (k1 < Ktot) {                     // stage next chunk into idle buffer
            stageA(cur ^ 1, k1);
            stageB(cur ^ 1, k1);
            int n = n0 + (tid & 127);        // prefetch chunk after next
            if (k1 + 32 < Ktot && n < Cout)
                __builtin_prefetch(&wgt[(long long)n * Ktot + k1 + 32], 0, 1);
        }

        // ---- fragment gather (ISA 16-bit A/B layouts), 2 x b128 per frag --
        v16h a0, a1, b0, b1;
        #pragma unroll
        for (int e = 0; e < 16; ++e) {
            int ka = ((e < 8) ? e : e + 8) + (half << 3);  // {0..7,16..23}/{8..15,24..31}
            a0[e] = Als[cur][mw + l16][ka];
            a1[e] = Als[cur][mw + 16 + l16][ka];
        }
        #pragma unroll
        for (int e = 0; e < 16; ++e) {
            int kb = ((e < 8) ? e : e + 8) + (half << 3);  // {0..7,16..23}/{8..15,24..31}
            b0[e] = Bls[cur][nw + l16][kb];
            b1[e] = Bls[cur][nw + 16 + l16][kb];
        }
        acc00 = __builtin_amdgcn_wmma_f32_16x16x32_f16(false, a0, false, b0, (short)0, acc00, false, false);
        acc01 = __builtin_amdgcn_wmma_f32_16x16x32_f16(false, a0, false, b1, (short)0, acc01, false, false);
        acc10 = __builtin_amdgcn_wmma_f32_16x16x32_f16(false, a1, false, b0, (short)0, acc10, false, false);
        acc11 = __builtin_amdgcn_wmma_f32_16x16x32_f16(false, a1, false, b1, (short)0, acc11, false, false);

        __syncthreads();   // closes reads of `cur` and writes of `cur^1`
    }

    // ---- epilogue: D lanes 0-15 N=lane M=r, lanes 16-31 N=lane-16 M=8+r ---
    #pragma unroll
    for (int ns = 0; ns < 2; ++ns) {
        int n = n0 + nw + (ns << 4) + l16;
        if (n >= Cout) continue;
        float bv = bias[n];
        float* outc = out + (long long)b * out_bstride + (long long)(out_coff + n) * HW;
        const float* resc = HAS_RES ? (res + ((long long)b * Cout + n) * HW) : nullptr;
        #pragma unroll
        for (int ms = 0; ms < 2; ++ms) {
            const v8f& a = (ns == 0) ? (ms == 0 ? acc00 : acc10)
                                     : (ms == 0 ? acc01 : acc11);
            #pragma unroll
            for (int r = 0; r < 8; ++r) {
                int pos = m0 + mw + (ms << 4) + r + (half << 3);
                float v = a[r] + bv;
                if (HAS_RES) v += resc[pos];
                outc[pos] = (v > 0.f) ? v : 0.2f * v;
            }
        }
    }
}

// ---------------------------------------------------------------------------
extern "C" void kernel_launch(void* const* d_in, const int* in_sizes, int n_in,
                              void* d_out, int out_size, void* d_ws, size_t ws_size,
                              hipStream_t stream) {
    (void)in_sizes; (void)n_in; (void)out_size; (void)ws_size;
    const int B = 8;
    const float* x   = (const float*)d_in[0];
    const float* w1w = (const float*)d_in[1];  const float* w1b = (const float*)d_in[2];
    const float* w2w = (const float*)d_in[3];  const float* w2b = (const float*)d_in[4];
    const float* w3w = (const float*)d_in[5];  const float* w3b = (const float*)d_in[6];
    const float* w4w = (const float*)d_in[7];  const float* w4b = (const float*)d_in[8];
    const float* d1w = (const float*)d_in[9];  const float* d1b = (const float*)d_in[10];
    const float* d2w = (const float*)d_in[11]; const float* d2b = (const float*)d_in[12];
    const float* d3w = (const float*)d_in[13]; const float* d3b = (const float*)d_in[14];
    const float* d4w = (const float*)d_in[15]; const float* d4b = (const float*)d_in[16];

    // ---- workspace carve (fp32 activations + f16 weights) ----
    float* ws = (float*)d_ws;
    size_t off = 0;
    auto alloc = [&](size_t n) { float* p = ws + off; off += n; return p; };
    float* w1   = alloc((size_t)B * 4    * 256 * 256);  // wt(x)
    float* cat2 = alloc((size_t)B * 32   * 256 * 256);  // [c1 | iwt(ic2)]
    float* w2   = alloc((size_t)B * 64   * 128 * 128);
    float* cat3 = alloc((size_t)B * 128  * 128 * 128);  // [c2 | iwt(ic3)]
    float* w3   = alloc((size_t)B * 256  * 64  * 64);
    float* cat4 = alloc((size_t)B * 512  * 64  * 64);   // [c3 | iwt(ic4)]
    float* w4   = alloc((size_t)B * 1024 * 32  * 32);
    float* c4   = alloc((size_t)B * 1024 * 32  * 32);
    float* c5   = alloc((size_t)B * 1024 * 32  * 32);
    float* ic4  = alloc((size_t)B * 1024 * 32  * 32);
    float* ic3  = alloc((size_t)B * 256  * 64  * 64);
    float* ic2  = alloc((size_t)B * 64   * 128 * 128);
    float* ic1  = alloc((size_t)B * 16   * 256 * 256);
    float* iw1  = alloc((size_t)B * 4    * 256 * 256);

    _Float16* hp = (_Float16*)(ws + off);
    size_t hoff = 0;
    auto halloc = [&](size_t n) { _Float16* p = hp + hoff; hoff += n; return p; };
    _Float16* h1  = halloc(16   * 4    * 9);
    _Float16* h2  = halloc(64   * 64   * 9);
    _Float16* h3  = halloc(256  * 256  * 9);
    _Float16* h4  = halloc(1024 * 1024 * 9);
    _Float16* hd1 = halloc(4    * 16   * 9);
    _Float16* hd2 = halloc(16   * 32   * 9);
    _Float16* hd3 = halloc(64   * 128  * 9);
    _Float16* hd4 = halloc(256  * 512  * 9);

    auto cvt = [&](const float* s, _Float16* d, int Cin, int n) {
        cvt_w_kernel<<<(n + 255) / 256, 256, 0, stream>>>(s, d, Cin, n);
    };
    cvt(w1w, h1, 4,    16*4*9);     cvt(w2w, h2, 64,   64*64*9);
    cvt(w3w, h3, 256,  256*256*9);  cvt(w4w, h4, 1024, 1024*1024*9);
    cvt(d1w, hd1, 16,  4*16*9);     cvt(d2w, hd2, 32,  16*32*9);
    cvt(d3w, hd3, 128, 64*128*9);   cvt(d4w, hd4, 512, 256*512*9);

    auto conv = [&](const float* in, long long inbs, const _Float16* wg, const float* bi,
                    float* out, long long obs, int coff, int lcin, int Cout, int lw,
                    const float* res) {
        dim3 grid(1 << (2 * lw - 6), (Cout + 127) >> 7, B);
        if (res)
            conv3x3_wmma_kernel<true, true><<<grid, 256, 0, stream>>>(in, inbs, wg, bi, res,
                out, obs, coff, lcin, Cout, lw);
        else if (lcin >= 5)
            conv3x3_wmma_kernel<false, true><<<grid, 256, 0, stream>>>(in, inbs, wg, bi, nullptr,
                out, obs, coff, lcin, Cout, lw);
        else
            conv3x3_wmma_kernel<false, false><<<grid, 256, 0, stream>>>(in, inbs, wg, bi, nullptr,
                out, obs, coff, lcin, Cout, lw);
    };
    auto wt = [&](const float* in, long long inbs, int coff, float* out, int lC, int lw) {
        int n = B << (lC + 2 * lw);
        wt_kernel<<<(n + 255) / 256, 256, 0, stream>>>(in, inbs, coff, out, lC, lw, n);
    };
    auto iwt = [&](const float* in, float* out, long long obs, int coff, int lC, int lw, bool sig) {
        int n = B << (lC + 2 * lw);
        if (sig) iwt_kernel<true ><<<(n + 255) / 256, 256, 0, stream>>>(in, out, obs, coff, lC, lw, n);
        else     iwt_kernel<false><<<(n + 255) / 256, 256, 0, stream>>>(in, out, obs, coff, lC, lw, n);
    };

    // ---------------- encoder ----------------
    wt(x, 512LL * 512, 0, w1, /*lC=*/0, /*lw=*/8);                        // w1 [8,4,256,256]
    conv(w1, 4LL*65536, h1, w1b, cat2, 32LL*65536, 0,   2,   16, 8, nullptr);   // c1
    wt(cat2, 32LL * 65536, 0, w2, 4, 7);                                  // w2 [8,64,128,128]
    conv(w2, 64LL*16384, h2, w2b, cat3, 128LL*16384, 0, 6,   64, 7, nullptr);   // c2
    wt(cat3, 128LL * 16384, 0, w3, 6, 6);                                 // w3 [8,256,64,64]
    conv(w3, 256LL*4096, h3, w3b, cat4, 512LL*4096, 0,  8,  256, 6, nullptr);   // c3
    wt(cat4, 512LL * 4096, 0, w4, 8, 5);                                  // w4 [8,1024,32,32]
    conv(w4, 1024LL*1024, h4, w4b, c4,  1024LL*1024, 0, 10, 1024, 5, nullptr);  // c4
    conv(c4, 1024LL*1024, h4, w4b, c5,  1024LL*1024, 0, 10, 1024, 5, nullptr);  // c5
    conv(c5, 1024LL*1024, h4, w4b, ic4, 1024LL*1024, 0, 10, 1024, 5, w4);       // ic4 = lrelu(c6+w4)

    // ---------------- decoder ----------------
    iwt(ic4, cat4, 512LL*4096, 256, 8, 5, false);                         // cat4[:,256:]
    conv(cat4, 512LL*4096, hd4, d4b, ic3, 256LL*4096, 0, 9, 256, 6, nullptr);   // ic3
    iwt(ic3, cat3, 128LL*16384, 64, 6, 6, false);                         // cat3[:,64:]
    conv(cat3, 128LL*16384, hd3, d3b, ic2, 64LL*16384, 0, 7, 64, 7, nullptr);   // ic2
    iwt(ic2, cat2, 32LL*65536, 16, 4, 7, false);                          // cat2[:,16:]
    conv(cat2, 32LL*65536, hd2, d2b, ic1, 16LL*65536, 0, 5, 16, 8, nullptr);    // ic1
    conv(ic1, 16LL*65536, hd1, d1b, iw1, 4LL*65536, 0,  4,  4, 8, nullptr);     // iw1
    iwt(iw1, (float*)d_out, 512LL*512, 0, 0, 8, true);                    // sigmoid(iwt)
}
